// TiedQuasiLSTM_89438398972450
// MI455X (gfx1250) — compile-verified, matrix operand order checked by reading, not run on hardware
//
#include <hip/hip_runtime.h>
#include <math.h>

// ---------------------------------------------------------------------------
// TiedQuasiLSTM for MI455X (gfx1250): bf16 WMMA GEMMs + chunked tied scan.
// v4: staging uses GLOBAL_LOAD_ASYNC_TO_LDS_B128 (ASYNCcnt-tracked direct
//     global->LDS copies, no VGPR round-trip), waited with s_wait_asynccnt
//     right before the per-iteration barrier. Compute reads fragments via
//     ds_load_b128 and issues v_wmma_f32_16x16x32_bf16.
// ---------------------------------------------------------------------------
#define DIMK      2048
#define NUM_HEADS 16
#define HEAD_DIM  128
#define HIDDEN    2048
#define BATCH     2
#define SEQ       4096
#define TOKENS    (BATCH * SEQ)     // 8192
#define NCHUNK    16
#define TCHUNK    (SEQ / NCHUNK)    // 256
#define BK        32                // K-step per WMMA

typedef __attribute__((ext_vector_type(16))) __bf16 v16bf;
typedef __attribute__((ext_vector_type(8)))  __bf16 v8bf;
typedef __attribute__((ext_vector_type(4)))  __bf16 v4bf;
typedef __attribute__((ext_vector_type(8)))  float  v8f;

__device__ __forceinline__ v16bf make_frag(v8bf lo, v8bf hi) {
  return __builtin_shufflevector(lo, hi, 0, 1, 2, 3, 4, 5, 6, 7,
                                 8, 9, 10, 11, 12, 13, 14, 15);
}

// Fragment from an LDS tile row-major with ld=BK(32) elements.
// ISA 7.12.2: lane L -> row (L&15); h=L>>4: elems 0..7 at K=8h+0..7,
// elems 8..15 at K=16+8h+0..7 -> 16B chunks at row byte offsets 16h, 32+16h.
__device__ __forceinline__ v16bf lds_frag(const __bf16* rowbase) {
  const int lane = threadIdx.x & 31;
  const __bf16* rp = rowbase + (size_t)(lane & 15) * BK + 8 * (lane >> 4);
  const v8bf lo = *reinterpret_cast<const v8bf*>(rp);
  const v8bf hi = *reinterpret_cast<const v8bf*>(rp + 16);
  return make_frag(lo, hi);
}

__device__ __forceinline__ v8f wmma_bf16(v16bf a, v16bf b, v8f c) {
  return __builtin_amdgcn_wmma_f32_16x16x32_bf16(false, a, false, b,
                                                 (short)0, c, false, false);
}

__device__ __forceinline__ float sigmoidf_(float x) {
  return 1.0f / (1.0f + __expf(-x));
}

// Async 16-byte global->LDS copy (per ISA 10 / 15.18.3, tracked by ASYNCcnt).
// LDS destination: low 32 bits of a generic pointer to __shared__ are the LDS
// byte offset (aperture scheme, ISA 10.2).
__device__ __forceinline__ void async_copy16(const __bf16* gsrc, __bf16* ldst) {
  const unsigned lds_off = (unsigned)(uintptr_t)ldst;
  asm volatile("global_load_async_to_lds_b128 %0, %1, off"
               :
               : "v"(lds_off), "v"(gsrc)
               : "memory");
}
__device__ __forceinline__ void wait_async0() {
  asm volatile("s_wait_asynccnt 0x0" ::: "memory");
}

// ---------------------------------------------------------------------------
// Kernel 0: one-shot fp32 -> bf16 conversion (vectorized 4 elems/thread).
// ---------------------------------------------------------------------------
__global__ void __launch_bounds__(256) cvt_f32_bf16_kernel(
    const float* __restrict__ src, __bf16* __restrict__ dst, size_t n) {
  const size_t i = ((size_t)blockIdx.x * 256 + threadIdx.x) * 4;
  if (i + 3 < n) {
    const float4 f = *reinterpret_cast<const float4*>(src + i);
    v4bf o;
    o[0] = (__bf16)f.x; o[1] = (__bf16)f.y; o[2] = (__bf16)f.z; o[3] = (__bf16)f.w;
    *reinterpret_cast<v4bf*>(dst + i) = o;
  }
}

// ---------------------------------------------------------------------------
// Kernel 1: fused gate GEMM for i/t/o blocks of g = x @ Wg^T + bg.
//   U[tok][c]  = sigmoid(g_i) * tanh(g_t)   (scan update, fp32)
//   Ob[tok][c] = g_o                        (output gate, bf16)
// Block = 256 thr (8 waves, 4Mx2N). Block tile M=128 x C=64 x 3 gates.
// LDS double buffer: A 2x128x32, B 2x192x32 bf16 (40 KB), async-staged.
// ---------------------------------------------------------------------------
__global__ void __launch_bounds__(256) gates_gemm_kernel(
    const __bf16* __restrict__ xb, const __bf16* __restrict__ Wgb,
    const float* __restrict__ bg, float* __restrict__ U,
    __bf16* __restrict__ Ob) {
  __shared__ __align__(16) __bf16 As[2][128][BK];
  __shared__ __align__(16) __bf16 Bs[2][192][BK];

  const int tid  = threadIdx.x;
  const int wave = tid >> 5;
  const int lane = tid & 31;
  const int mblk = blockIdx.y * 128;
  const int cblk = blockIdx.x * 64;
  const int mw = wave & 3, nw = wave >> 2;

  v8f acc[3][2][2];
#pragma unroll
  for (int g = 0; g < 3; ++g)
#pragma unroll
    for (int mi = 0; mi < 2; ++mi)
#pragma unroll
      for (int ni = 0; ni < 2; ++ni) acc[g][mi][ni] = (v8f)0.0f;

  // Cooperative async global->LDS staging of one K-step tile pair.
  auto stage = [&](int kt, int buf) {
    const int k = kt * BK;
    // A tile: 128x32 bf16 = 512 x 16B chunks, 2 per thread.
#pragma unroll
    for (int j = 0; j < 2; ++j) {
      const int chunk = tid + 256 * j;      // 0..511
      const int row = chunk >> 2, q = chunk & 3;
      const __bf16* src = xb + (size_t)(mblk + row) * DIMK + k + 8 * q;
      if (j == 0) __builtin_prefetch(src + BK, 0, 1);  // next-next K tile
      async_copy16(src, &As[buf][row][8 * q]);
    }
    // B tile: 192x32 bf16 = 768 x 16B chunks, 3 per thread. Row g*64+n maps
    // to Wg row 16 + g*HIDDEN + cblk + n.
#pragma unroll
    for (int j = 0; j < 3; ++j) {
      const int chunk = tid + 256 * j;      // 0..767
      const int row = chunk >> 2, q = chunk & 3;
      const int g = row >> 6, n = row & 63;
      const __bf16* src =
          Wgb + (size_t)(NUM_HEADS + g * HIDDEN + cblk + n) * DIMK + k + 8 * q;
      if (j == 0) __builtin_prefetch(src + BK, 0, 1);
      async_copy16(src, &Bs[buf][row][8 * q]);
    }
  };

  stage(0, 0);
  wait_async0();
  __syncthreads();

  const int NK = DIMK / BK;
  for (int kt = 0; kt < NK; ++kt) {
    const int buf = kt & 1;
    if (kt + 1 < NK) stage(kt + 1, buf ^ 1);  // async; lands during compute

    const v16bf a0 = lds_frag(&As[buf][mw * 32][0]);
    const v16bf a1 = lds_frag(&As[buf][mw * 32 + 16][0]);
#pragma unroll
    for (int g = 0; g < 3; ++g) {
      const v16bf b0 = lds_frag(&Bs[buf][g * 64 + nw * 32][0]);
      const v16bf b1 = lds_frag(&Bs[buf][g * 64 + nw * 32 + 16][0]);
      acc[g][0][0] = wmma_bf16(a0, b0, acc[g][0][0]);
      acc[g][1][0] = wmma_bf16(a1, b0, acc[g][1][0]);
      acc[g][0][1] = wmma_bf16(a0, b1, acc[g][0][1]);
      acc[g][1][1] = wmma_bf16(a1, b1, acc[g][1][1]);
    }
    wait_async0();      // staged tile landed
    __syncthreads();    // all waves done reading buf & staging complete
  }

  const int col   = lane & 15;
  const int rbase = (lane >> 4) * 8;
  const int m0 = mblk + mw * 32;
  const int c0 = cblk + nw * 32;
#pragma unroll
  for (int mi = 0; mi < 2; ++mi) {
#pragma unroll
    for (int ni = 0; ni < 2; ++ni) {
      const int c = c0 + ni * 16 + col;
      const float bi  = bg[NUM_HEADS + c];
      const float bt  = bg[NUM_HEADS + HIDDEN + c];
      const float bo_ = bg[NUM_HEADS + 2 * HIDDEN + c];
#pragma unroll
      for (int v = 0; v < 8; ++v) {
        const size_t tok = (size_t)(m0 + mi * 16 + rbase + v);
        const float gi = acc[0][mi][ni][v] + bi;
        const float gt = acc[1][mi][ni][v] + bt;
        const float go = acc[2][mi][ni][v] + bo_;
        U[tok * DIMK + c]  = sigmoidf_(gi) * tanhf(gt);
        Ob[tok * DIMK + c] = (__bf16)go;
      }
    }
  }
}

// ---------------------------------------------------------------------------
// Kernel 2: tied forget-gate GEMM (16 heads): Fh[tok][h] = sigmoid(x@Wf^T+bf).
// Tiny (N=16); register-direct fragments are fine.
// ---------------------------------------------------------------------------
__global__ void __launch_bounds__(256) fgate_gemm_kernel(
    const __bf16* __restrict__ xb, const __bf16* __restrict__ Wgb,
    const float* __restrict__ bg, float* __restrict__ Fh) {
  const int wave = threadIdx.x >> 5;
  const int lane = threadIdx.x & 31;
  const int m0 = blockIdx.x * 128 + wave * 16;
  const int r = lane & 15, h2 = lane >> 4;

  v8f acc = (v8f)0.0f;
  for (int k = 0; k < DIMK; k += BK) {
    const __bf16* ap = xb + (size_t)(m0 + r) * DIMK + k + 8 * h2;
    const __bf16* bp = Wgb + (size_t)r * DIMK + k + 8 * h2;  // rows 0..15 = f
    const v16bf a = make_frag(*reinterpret_cast<const v8bf*>(ap),
                              *reinterpret_cast<const v8bf*>(ap + 16));
    const v16bf b = make_frag(*reinterpret_cast<const v8bf*>(bp),
                              *reinterpret_cast<const v8bf*>(bp + 16));
    acc = wmma_bf16(a, b, acc);
  }
  const int h     = lane & 15;
  const int rbase = (lane >> 4) * 8;
  const float bh  = bg[h];
#pragma unroll
  for (int v = 0; v < 8; ++v) {
    const size_t tok = (size_t)(m0 + rbase + v);
    Fh[tok * NUM_HEADS + h] = sigmoidf_(acc[v] + bh);
  }
}

// ---------------------------------------------------------------------------
// Kernel 3: per-chunk local scan. U rewritten in-place with local scan c^loc.
// Cumulative forget products stored per head (tied across HEAD_DIM).
// ---------------------------------------------------------------------------
__global__ void __launch_bounds__(256) scan_chunk_kernel(
    float* __restrict__ U /* in: update, out: local c */,
    const float* __restrict__ Fh, float* __restrict__ Fcumh,
    float* __restrict__ Fchunk, float* __restrict__ Clast) {
  const int bid   = blockIdx.x;
  const int cblk  = bid & 7;
  const int chunk = (bid >> 3) & (NCHUNK - 1);
  const int b     = bid >> 7;
  const int c     = cblk * 256 + threadIdx.x;
  const int h     = c >> 7;  // HEAD_DIM = 128
  const size_t tok0 = (size_t)b * SEQ + (size_t)chunk * TCHUNK;

  float cloc = 0.0f, fprod = 1.0f;
  for (int t = 0; t < TCHUNK; ++t) {
    const size_t tok = tok0 + t;
    const float f = Fh[tok * NUM_HEADS + h];
    const float u = U[tok * DIMK + c];
    cloc  = fmaf(f, cloc, u);
    fprod *= f;
    U[tok * DIMK + c] = cloc;
    if ((c & (HEAD_DIM - 1)) == 0) Fcumh[tok * NUM_HEADS + h] = fprod;
  }
  const size_t ck = (size_t)(b * NCHUNK + chunk);
  Clast[ck * DIMK + c] = cloc;
  if ((c & (HEAD_DIM - 1)) == 0) Fchunk[ck * NUM_HEADS + h] = fprod;
}

// ---------------------------------------------------------------------------
// Kernel 4: tiny sequential carry scan across the 16 chunks (per channel).
// ---------------------------------------------------------------------------
__global__ void __launch_bounds__(256) scan_carry_kernel(
    const float* __restrict__ Fchunk, const float* __restrict__ Clast,
    float* __restrict__ Carry) {
  const int idx = blockIdx.x * 256 + threadIdx.x;  // 0 .. BATCH*DIMK-1
  const int b = idx >> 11;
  const int c = idx & (DIMK - 1);
  const int h = c >> 7;
  float carry = 0.0f;
  for (int k = 0; k < NCHUNK; ++k) {
    const size_t ck = (size_t)(b * NCHUNK + k);
    Carry[ck * DIMK + c] = carry;
    carry = fmaf(Fchunk[ck * NUM_HEADS + h], carry, Clast[ck * DIMK + c]);
  }
}

// ---------------------------------------------------------------------------
// Kernel 5: apply carries + output gate: Hb = Ob * tanh(c^loc + fcum*carry).
// ---------------------------------------------------------------------------
__global__ void __launch_bounds__(256) apply_kernel(
    const float* __restrict__ Cloc, const __bf16* __restrict__ Ob,
    const float* __restrict__ Fcumh, const float* __restrict__ Carry,
    __bf16* __restrict__ Hb) {
  const size_t idx = (size_t)blockIdx.x * 256 + threadIdx.x;
  const size_t tok = idx / DIMK;
  const int    c   = (int)(idx & (DIMK - 1));
  const int    b   = (int)(tok >> 12);           // tok / SEQ
  const int    t   = (int)(tok & (SEQ - 1));
  const int    chunk = t >> 8;                   // t / TCHUNK
  const int    h   = c >> 7;
  const size_t ck  = (size_t)(b * NCHUNK + chunk);
  const float cfull = fmaf(Fcumh[tok * NUM_HEADS + h], Carry[ck * DIMK + c],
                           Cloc[idx]);
  Hb[idx] = (__bf16)((float)Ob[idx] * tanhf(cfull));
}

// ---------------------------------------------------------------------------
// Kernel 6: output GEMM: out = H @ Wo^T + bo. LDS double buffer (24 KB),
// async-staged.
// ---------------------------------------------------------------------------
__global__ void __launch_bounds__(256) out_gemm_kernel(
    const __bf16* __restrict__ Hb, const __bf16* __restrict__ Wob,
    const float* __restrict__ bo, float* __restrict__ out) {
  __shared__ __align__(16) __bf16 As[2][128][BK];
  __shared__ __align__(16) __bf16 Bs[2][64][BK];

  const int tid  = threadIdx.x;
  const int wave = tid >> 5;
  const int lane = tid & 31;
  const int mblk = blockIdx.y * 128;
  const int dblk = blockIdx.x * 64;
  const int mw = wave & 3, nw = wave >> 2;

  v8f acc[2][2];
#pragma unroll
  for (int mi = 0; mi < 2; ++mi)
#pragma unroll
    for (int ni = 0; ni < 2; ++ni) acc[mi][ni] = (v8f)0.0f;

  auto stage = [&](int kt, int buf) {
    const int k = kt * BK;
#pragma unroll
    for (int j = 0; j < 2; ++j) {               // A: 512 chunks, 2/thread
      const int chunk = tid + 256 * j;
      const int row = chunk >> 2, q = chunk & 3;
      const __bf16* src = Hb + (size_t)(mblk + row) * HIDDEN + k + 8 * q;
      if (j == 0) __builtin_prefetch(src + BK, 0, 1);
      async_copy16(src, &As[buf][row][8 * q]);
    }
    {                                            // B: 256 chunks, 1/thread
      const int row = tid >> 2, q = tid & 3;
      const __bf16* src = Wob + (size_t)(dblk + row) * HIDDEN + k + 8 * q;
      __builtin_prefetch(src + BK, 0, 1);
      async_copy16(src, &Bs[buf][row][8 * q]);
    }
  };

  stage(0, 0);
  wait_async0();
  __syncthreads();

  const int NK = HIDDEN / BK;
  for (int kt = 0; kt < NK; ++kt) {
    const int buf = kt & 1;
    if (kt + 1 < NK) stage(kt + 1, buf ^ 1);

    const v16bf a0 = lds_frag(&As[buf][mw * 32][0]);
    const v16bf a1 = lds_frag(&As[buf][mw * 32 + 16][0]);
    const v16bf b0 = lds_frag(&Bs[buf][nw * 32][0]);
    const v16bf b1 = lds_frag(&Bs[buf][nw * 32 + 16][0]);
    acc[0][0] = wmma_bf16(a0, b0, acc[0][0]);
    acc[1][0] = wmma_bf16(a1, b0, acc[1][0]);
    acc[0][1] = wmma_bf16(a0, b1, acc[0][1]);
    acc[1][1] = wmma_bf16(a1, b1, acc[1][1]);
    wait_async0();
    __syncthreads();
  }

  const int col   = lane & 15;
  const int rbase = (lane >> 4) * 8;
  const int m0 = mblk + mw * 32;
  const int d0 = dblk + nw * 32;
#pragma unroll
  for (int mi = 0; mi < 2; ++mi) {
#pragma unroll
    for (int ni = 0; ni < 2; ++ni) {
      const int d = d0 + ni * 16 + col;
      const float bd = bo[d];
#pragma unroll
      for (int v = 0; v < 8; ++v) {
        const size_t tok = (size_t)(m0 + mi * 16 + rbase + v);
        out[tok * DIMK + d] = acc[mi][ni][v] + bd;
      }
    }
  }
}

// ---------------------------------------------------------------------------
extern "C" void kernel_launch(void* const* d_in, const int* in_sizes, int n_in,
                              void* d_out, int out_size, void* d_ws, size_t ws_size,
                              hipStream_t stream) {
  const float* x  = (const float*)d_in[0];   // (2, 4096, 2048)
  const float* Wg = (const float*)d_in[1];   // (6160, 2048)
  const float* bg = (const float*)d_in[2];   // (6160,)
  const float* Wo = (const float*)d_in[3];   // (2048, 2048)
  const float* bo = (const float*)d_in[4];   // (2048,)
  float* out = (float*)d_out;                // (2, 4096, 2048)

  const size_t NX  = (size_t)TOKENS * DIMK;                    // 16.8M
  const size_t NWG = (size_t)(NUM_HEADS + 3 * HIDDEN) * DIMK;  // 12.6M
  const size_t NWO = (size_t)DIMK * HIDDEN;                    // 4.2M

  // Workspace carve-up (~203 MB total).
  char* cur = (char*)d_ws;
  float* U      = (float*)cur; cur += NX * sizeof(float);      // update -> local c
  float* Fh     = (float*)cur; cur += (size_t)TOKENS * NUM_HEADS * sizeof(float);
  float* Fcumh  = (float*)cur; cur += (size_t)TOKENS * NUM_HEADS * sizeof(float);
  float* Fchunk = (float*)cur; cur += (size_t)BATCH * NCHUNK * NUM_HEADS * sizeof(float);
  float* Clast  = (float*)cur; cur += (size_t)BATCH * NCHUNK * DIMK * sizeof(float);
  float* Carry  = (float*)cur; cur += (size_t)BATCH * NCHUNK * DIMK * sizeof(float);
  __bf16* xb    = (__bf16*)cur; cur += NX  * sizeof(__bf16);
  __bf16* Wgb   = (__bf16*)cur; cur += NWG * sizeof(__bf16);
  __bf16* Wob   = (__bf16*)cur; cur += NWO * sizeof(__bf16);
  __bf16* Ob    = (__bf16*)cur; cur += NX  * sizeof(__bf16);
  __bf16* Hb    = (__bf16*)cur; cur += NX  * sizeof(__bf16);

  // 0. One-shot bf16 conversions of GEMM operands.
  cvt_f32_bf16_kernel<<<(int)(NX  / 4 / 256), 256, 0, stream>>>(x,  xb,  NX);
  cvt_f32_bf16_kernel<<<(int)((NWG / 4 + 255) / 256), 256, 0, stream>>>(Wg, Wgb, NWG);
  cvt_f32_bf16_kernel<<<(int)(NWO / 4 / 256), 256, 0, stream>>>(Wo, Wob, NWO);

  // 1. Gate GEMM (i/t/o fused) and 2. tied forget-gate GEMM.
  gates_gemm_kernel<<<dim3(DIMK / 64, TOKENS / 128), 256, 0, stream>>>(xb, Wgb, bg, U, Ob);
  fgate_gemm_kernel<<<TOKENS / 128, 256, 0, stream>>>(xb, Wgb, bg, Fh);

  // 3-5. Chunked associative scan over time.
  scan_chunk_kernel<<<BATCH * NCHUNK * (DIMK / 256), 256, 0, stream>>>(U, Fh, Fcumh, Fchunk, Clast);
  scan_carry_kernel<<<(BATCH * DIMK) / 256, 256, 0, stream>>>(Fchunk, Clast, Carry);
  apply_kernel<<<(int)(NX / 256), 256, 0, stream>>>(U, Ob, Fcumh, Carry, Hb);

  // 6. Output projection.
  out_gemm_kernel<<<dim3(DIMK / 64, TOKENS / 128), 256, 0, stream>>>(Hb, Wob, bo, out);
}
